// LengthRegulator_8375186227764
// MI455X (gfx1250) — matrix operand order repaired and checked by reading, not compiled, not run on hardware
//
#include <hip/hip_runtime.h>

// ---------------------------------------------------------------------------
// Problem constants (from reference)
// ---------------------------------------------------------------------------
#define BB   16
#define LL   4096
#define CC   384     // channels everywhere (C == H == 384)
#define T1   4094    // L - (K-1)
#define T2   4092    // L - 2*(K-1)
#define LN_EPS 1e-5f

typedef __attribute__((ext_vector_type(16))) __bf16 v16bf;
typedef __attribute__((ext_vector_type(8)))  __bf16 v8bf;
typedef __attribute__((ext_vector_type(8)))  float  v8f;
typedef __attribute__((ext_vector_type(4)))  unsigned v4u;
typedef __attribute__((ext_vector_type(8)))  int      v8i;
typedef __attribute__((ext_vector_type(4)))  int      v4i;

__device__ __forceinline__ __bf16 to_bf16(float f) {
    unsigned u = __builtin_bit_cast(unsigned, f);
    u += 0x7FFFu + ((u >> 16) & 1u);            // round-to-nearest-even
    unsigned short h = (unsigned short)(u >> 16);
    return __builtin_bit_cast(__bf16, h);
}

// ---------------------------------------------------------------------------
// fp32 -> bf16 bulk convert (8 elements / thread)
// ---------------------------------------------------------------------------
__global__ void cvt_bf16_kernel(const float* __restrict__ src,
                                __bf16* __restrict__ dst, long long n8) {
    long long i = (long long)blockIdx.x * blockDim.x + threadIdx.x;
    if (i >= n8) return;
    const float4* s = (const float4*)src + 2 * i;
    float4 a = s[0], b = s[1];
    __bf16* d = dst + 8 * i;
    d[0] = to_bf16(a.x); d[1] = to_bf16(a.y); d[2] = to_bf16(a.z); d[3] = to_bf16(a.w);
    d[4] = to_bf16(b.x); d[5] = to_bf16(b.y); d[6] = to_bf16(b.z); d[7] = to_bf16(b.w);
}

// ---------------------------------------------------------------------------
// Pack conv weights (H,C,K) fp32 into WMMA B-operand tiles, bf16.
// Tile (kt, cc, ht): 32x16 (KxN). Per lane: 16 contiguous halves
//   lane n (0..15):  column h0+n,  K = c0+0..15
//   lane n+16:       column h0+n,  K = c0+16..31
// grid = 3*12*24 tiles, block = 32 (one lane each)
// ---------------------------------------------------------------------------
__global__ void pack_w_kernel(const float* __restrict__ w, __bf16* __restrict__ wpk) {
    int tile = blockIdx.x;
    int lane = threadIdx.x;
    int kt = tile / (12 * 24);
    int rem = tile % (12 * 24);
    int ccu = rem / 24;
    int ht  = rem % 24;
    int col = lane & 15;
    int kb  = (lane >> 4) * 16;
    int h   = ht * 16 + col;
    __bf16* out = wpk + (size_t)tile * 512 + lane * 16;
#pragma unroll
    for (int j = 0; j < 16; ++j) {
        int c = ccu * 32 + kb + j;
        out[j] = to_bf16(w[((size_t)h * CC + c) * 3 + kt]);   // w[h, c, kt]
    }
}

// ---------------------------------------------------------------------------
// K=3 stride-1 VALID conv over channels as 3 shifted bf16 WMMA GEMMs.
// Workgroup: 256 threads (8 wave32). Tile: M=64 time steps x N=384 channels.
// A staged into LDS by the Tensor Data Mover (one 384x66 bf16 2D tile;
// rows past Tin zero-fill via tensor_dim1 OOB). B from packed global (L2).
// Output: raw fp32 conv + bias, [B, Tout, 384].
// ---------------------------------------------------------------------------
__global__ __launch_bounds__(256)
void conv_wmma_kernel(const __bf16* __restrict__ in,   // [B, Tin, 384]
                      const __bf16* __restrict__ wpk,  // packed tiles
                      const float*  __restrict__ bias, // [384]
                      float* __restrict__ out,         // [B, Tout, 384]
                      int Tin, int Tout) {
    __shared__ __align__(16) __bf16 smem[66 * CC];

    const int b   = blockIdx.y;
    const int m0  = blockIdx.x * 64;
    const int tid = threadIdx.x;
    const size_t inb = (size_t)b * Tin * CC;

    const int wave  = tid >> 5;       // 0..7, owns h-tiles wave*3 .. wave*3+2
    const int lane  = tid & 31;
    const int lhalf = lane >> 4;      // 0|1
    const int l15   = lane & 15;

    // warm L2 with this wave's packed-weight slice (global_prefetch_b8)
    __builtin_prefetch(wpk + (size_t)(wave * 3) * 512 + lane * 16, 0, 0);

#if __has_builtin(__builtin_amdgcn_tensor_load_to_lds)
    // ---- TDM: DMA a 384 x 66 bf16 tile (row stride 384) into LDS ----
    if (wave == 0) {
        unsigned long long ga =
            (unsigned long long)(const void*)(in + inb + (size_t)m0 * CC);
        unsigned lds_off = (unsigned)(unsigned long long)(void*)smem;
        unsigned rows = (unsigned)(Tin - m0);   // tensor_dim1: OOB rows read 0
        v4u g0 = { 1u,                                   // count=1, user mode
                   lds_off,                              // lds_addr
                   (unsigned)ga,                         // global_addr[31:0]
                   (((unsigned)(ga >> 32)) & 0x01FFFFFFu) | (2u << 30) }; // addr[56:32] | type=2
        v8i g1 = { (int)(1u << 16),                      // data_size=2B, mask=0
                   (int)(384u << 16),                    // tensor_dim0[15:0]=384
                   (int)((rows & 0xFFFFu) << 16),        // dim0[31:16]=0 | dim1[15:0]
                   (int)((384u << 16) | ((rows >> 16) & 0xFFFFu)), // dim1[31:16] | tile_dim0=384
                   66,                                   // tile_dim1=66, tile_dim2=0
                   384,                                  // tensor_dim0_stride[31:0]
                   0, 0 };                               // stride hi / dim1_stride
        v4i gz = { 0, 0, 0, 0 };
        v8i gz8 = { 0, 0, 0, 0, 0, 0, 0, 0 };
        __builtin_amdgcn_tensor_load_to_lds(g0, g1, gz, gz, gz8, 0);
        __builtin_amdgcn_s_wait_tensorcnt(0);
    }
#else
    // fallback: cooperative staging
    for (int u = tid; u < 66 * 48; u += 256) {
        int r = u / 48, cu = u % 48;
        int rg = m0 + r; if (rg > Tin - 1) rg = Tin - 1;
        *((uint4*)(smem + r * CC) + cu) =
            *((const uint4*)(in + inb + (size_t)rg * CC) + cu);
    }
#endif
    __syncthreads();

    v8f acc[4][3] = {};

    for (int kt = 0; kt < 3; ++kt) {
        for (int ccu = 0; ccu < 12; ++ccu) {
            // B tiles: 32 contiguous bytes per lane from packed weights
            v16bf bt[3];
#pragma unroll
            for (int j = 0; j < 3; ++j) {
                const __bf16* p = wpk +
                    (size_t)(((kt * 12 + ccu) * 24) + (wave * 3 + j)) * 512 + lane * 16;
                bt[j] = *(const v16bf*)p;
            }
            // A tiles from LDS: two b128 reads per lane, contiguous K = 16*lhalf + 0..15
            v16bf at[4];
#pragma unroll
            for (int mt = 0; mt < 4; ++mt) {
                int row = mt * 16 + l15 + kt;                 // tap shift
                const __bf16* p = smem + row * CC + ccu * 32 + lhalf * 16;
                v8bf lo = *(const v8bf*)p;
                v8bf hi = *(const v8bf*)(p + 8);
                at[mt] = __builtin_shufflevector(lo, hi,
                          0, 1, 2, 3, 4, 5, 6, 7, 8, 9, 10, 11, 12, 13, 14, 15);
            }
#pragma unroll
            for (int mt = 0; mt < 4; ++mt)
#pragma unroll
                for (int j = 0; j < 3; ++j)
                    acc[mt][j] = __builtin_amdgcn_wmma_f32_16x16x32_bf16(
                        false, at[mt], false, bt[j], (short)0, acc[mt][j], false, false);
        }
    }

    // ---- epilogue: + bias, write fp32 raw ----
    const size_t outb = (size_t)b * Tout * CC;
    const bool full = (m0 + 64) <= Tout;
#pragma unroll
    for (int j = 0; j < 3; ++j) {
        int h = (wave * 3 + j) * 16 + l15;
        float bv = bias[h];
        float* op = out + outb + h;
#pragma unroll
        for (int mt = 0; mt < 4; ++mt) {
            int tbase = m0 + mt * 16 + 8 * lhalf;
            if (full) {
#pragma unroll
                for (int v = 0; v < 8; ++v)
                    op[(size_t)(tbase + v) * CC] = acc[mt][j][v] + bv;
            } else {
#pragma unroll
                for (int v = 0; v < 8; ++v) {
                    int t = tbase + v;
                    if (t < Tout) op[(size_t)t * CC] = acc[mt][j][v] + bv;
                }
            }
        }
    }
}

// ---------------------------------------------------------------------------
// LayerNorm(384) + ReLU -> bf16. One 128-thread block per row, 3 elems/thread.
// ---------------------------------------------------------------------------
__global__ __launch_bounds__(128)
void ln_relu_bf16_kernel(const float* __restrict__ raw,
                         const float* __restrict__ g, const float* __restrict__ bv,
                         __bf16* __restrict__ out) {
    __shared__ float s1[128], s2[128];
    const size_t row = blockIdx.x;
    const int tid = threadIdx.x;
    const float* r = raw + row * CC;
    float x0 = r[tid], x1 = r[tid + 128], x2 = r[tid + 256];
    s1[tid] = x0 + x1 + x2;
    s2[tid] = x0 * x0 + x1 * x1 + x2 * x2;
    __syncthreads();
    for (int off = 64; off > 0; off >>= 1) {
        if (tid < off) { s1[tid] += s1[tid + off]; s2[tid] += s2[tid + off]; }
        __syncthreads();
    }
    float mean = s1[0] * (1.0f / CC);
    float var  = s2[0] * (1.0f / CC) - mean * mean;
    float rs   = rsqrtf(var + LN_EPS);
    __bf16* o = out + row * CC;
    float y;
    y = (x0 - mean) * rs * g[tid]       + bv[tid];       o[tid]       = to_bf16(fmaxf(y, 0.f));
    y = (x1 - mean) * rs * g[tid + 128] + bv[tid + 128]; o[tid + 128] = to_bf16(fmaxf(y, 0.f));
    y = (x2 - mean) * rs * g[tid + 256] + bv[tid + 256]; o[tid + 256] = to_bf16(fmaxf(y, 0.f));
}

// ---------------------------------------------------------------------------
// LayerNorm + ReLU + Linear(384->1) + exp, fused. One block per row.
// ---------------------------------------------------------------------------
__global__ __launch_bounds__(128)
void ln_lin_exp_kernel(const float* __restrict__ raw,
                       const float* __restrict__ g, const float* __restrict__ bv,
                       const float* __restrict__ lw, const float* __restrict__ lb,
                       float* __restrict__ dur) {
    __shared__ float s1[128], s2[128];
    const size_t row = blockIdx.x;
    const int tid = threadIdx.x;
    const float* r = raw + row * CC;
    float x0 = r[tid], x1 = r[tid + 128], x2 = r[tid + 256];
    s1[tid] = x0 + x1 + x2;
    s2[tid] = x0 * x0 + x1 * x1 + x2 * x2;
    __syncthreads();
    for (int off = 64; off > 0; off >>= 1) {
        if (tid < off) { s1[tid] += s1[tid + off]; s2[tid] += s2[tid + off]; }
        __syncthreads();
    }
    float mean = s1[0] * (1.0f / CC);
    float var  = s2[0] * (1.0f / CC) - mean * mean;
    float rs   = rsqrtf(var + LN_EPS);
    float y0 = fmaxf((x0 - mean) * rs * g[tid]       + bv[tid],       0.f);
    float y1 = fmaxf((x1 - mean) * rs * g[tid + 128] + bv[tid + 128], 0.f);
    float y2 = fmaxf((x2 - mean) * rs * g[tid + 256] + bv[tid + 256], 0.f);
    float part = y0 * lw[tid] + y1 * lw[tid + 128] + y2 * lw[tid + 256];
    __syncthreads();
    s1[tid] = part;
    __syncthreads();
    for (int off = 64; off > 0; off >>= 1) {
        if (tid < off) s1[tid] += s1[tid + off];
        __syncthreads();
    }
    if (tid == 0) dur[row] = __expf(s1[0] + lb[0]);
}

// ---------------------------------------------------------------------------
// Per-batch inclusive cumsum of target durations. One block per batch.
// ---------------------------------------------------------------------------
__global__ __launch_bounds__(256)
void cumsum_kernel(const int* __restrict__ tgt, int* __restrict__ ends) {
    __shared__ int part[256];
    const int b = blockIdx.x, tid = threadIdx.x;
    const int* tb = tgt + (size_t)b * LL;
    int* eb = ends + (size_t)b * LL;
    int base = tid * 16;
    int s = 0;
#pragma unroll
    for (int i = 0; i < 16; ++i) s += tb[base + i];
    part[tid] = s;
    __syncthreads();
    if (tid == 0) {
        int run = 0;
        for (int i = 0; i < 256; ++i) { int p = part[i]; part[i] = run; run += p; }
    }
    __syncthreads();
    int run = part[tid];
#pragma unroll
    for (int i = 0; i < 16; ++i) { run += tb[base + i]; eb[base + i] = run; }
}

// ---------------------------------------------------------------------------
// Length regulator: one 96-thread block (3 x wave32) per output frame.
// Binary search cumsum, copy 384 floats as float4 (zero past total duration).
// ---------------------------------------------------------------------------
__global__ __launch_bounds__(96)
void regulate_kernel(const float* __restrict__ x, const int* __restrict__ ends,
                     float* __restrict__ out, int T) {
    const int blk = blockIdx.x;
    const int b = blk / T;
    const int t = blk % T;
    const int* e = ends + (size_t)b * LL;
    int lo = 0, hi = LL;
    while (lo < hi) { int mid = (lo + hi) >> 1; if (e[mid] <= t) lo = mid + 1; else hi = mid; }
    bool valid = lo < LL;
    int src = valid ? lo : (LL - 1);
    float4* o = (float4*)(out + (size_t)blk * CC);
    if (valid) {
        const float4* s = (const float4*)(x + ((size_t)b * LL + src) * CC);
        o[threadIdx.x] = s[threadIdx.x];
    } else {
        o[threadIdx.x] = make_float4(0.f, 0.f, 0.f, 0.f);
    }
}

// ---------------------------------------------------------------------------
// Host launcher
// ---------------------------------------------------------------------------
extern "C" void kernel_launch(void* const* d_in, const int* in_sizes, int n_in,
                              void* d_out, int out_size, void* d_ws, size_t ws_size,
                              hipStream_t stream) {
    const float* x       = (const float*)d_in[0];
    const int*   target  = (const int*)  d_in[1];
    const float* conv1_w = (const float*)d_in[2];
    const float* conv1_b = (const float*)d_in[3];
    const float* ln1_g   = (const float*)d_in[4];
    const float* ln1_b   = (const float*)d_in[5];
    const float* conv2_w = (const float*)d_in[6];
    const float* conv2_b = (const float*)d_in[7];
    const float* ln2_g   = (const float*)d_in[8];
    const float* ln2_b   = (const float*)d_in[9];
    const float* lin_w   = (const float*)d_in[10];
    const float* lin_b   = (const float*)d_in[11];

    // derive data-dependent T from out_size: d_out = [B*T*C output | B*T2 durations]
    long long durN = (long long)BB * T2;
    long long T = ((long long)out_size - durN) / ((long long)BB * CC);
    if (T < 0) T = 0;

    // workspace carve-up
    char* w = (char*)d_ws;
    float*  raw  = (float*)w;  w += (size_t)BB * T1 * CC * sizeof(float);   // reused conv1/conv2
    __bf16* xb   = (__bf16*)w; w += (size_t)BB * LL * CC * sizeof(__bf16);
    __bf16* h1   = (__bf16*)w; w += (size_t)BB * T1 * CC * sizeof(__bf16);
    __bf16* w1p  = (__bf16*)w; w += (size_t)3 * CC * CC * sizeof(__bf16);
    __bf16* w2p  = (__bf16*)w; w += (size_t)3 * CC * CC * sizeof(__bf16);
    int*    ends = (int*)w;    w += (size_t)BB * LL * sizeof(int);

    // 1. x -> bf16
    long long n8 = (long long)BB * LL * CC / 8;
    cvt_bf16_kernel<<<(unsigned)((n8 + 255) / 256), 256, 0, stream>>>(x, xb, n8);

    // 2. pack weights into WMMA B-operand tiles
    pack_w_kernel<<<3 * 12 * 24, 32, 0, stream>>>(conv1_w, w1p);
    pack_w_kernel<<<3 * 12 * 24, 32, 0, stream>>>(conv2_w, w2p);

    // 3. conv1 (WMMA + TDM staging) -> raw fp32
    conv_wmma_kernel<<<dim3((T1 + 63) / 64, BB), 256, 0, stream>>>(
        xb, w1p, conv1_b, raw, LL, T1);

    // 4. LN1 + ReLU -> h1 bf16
    ln_relu_bf16_kernel<<<BB * T1, 128, 0, stream>>>(raw, ln1_g, ln1_b, h1);

    // 5. conv2 (WMMA + TDM staging) -> raw fp32 (buffer reused)
    conv_wmma_kernel<<<dim3((T2 + 63) / 64, BB), 256, 0, stream>>>(
        h1, w2p, conv2_b, raw, T1, T2);

    // 6. LN2 + ReLU + linear + exp -> durations (second output region)
    float* dur = (float*)d_out + (size_t)BB * T * CC;
    ln_lin_exp_kernel<<<BB * T2, 128, 0, stream>>>(raw, ln2_g, ln2_b, lin_w, lin_b, dur);

    // 7. cumsum of target durations
    cumsum_kernel<<<BB, 256, 0, stream>>>(target, ends);

    // 8. length regulate -> first output region
    if (T > 0) {
        regulate_kernel<<<(unsigned)(BB * T), 96, 0, stream>>>(
            x, ends, (float*)d_out, (int)T);
    }
}
